// TopDownTreeLSTM_33792802685045
// MI455X (gfx1250) — matrix operand order repaired and checked by reading, not compiled, not run on hardware
//
#include <hip/hip_runtime.h>

#define TD_D 768
#define TD_N 32767
#define TD_LEVELS 15
#define LDS_STRIDE 776      // 768 + 8 pad: conflict-free ds_load_b128 across rows
#define WROWS 3072          // combined rows: [i(768); o(768); u(768); f(768)]

typedef __attribute__((ext_vector_type(16))) __bf16 v16bf;
typedef __attribute__((ext_vector_type(8)))  float  v8f;

// ---------- bf16 split helpers (hi = RNE(x), lo = RNE(x - hi)) ----------
static __device__ __forceinline__ unsigned short bf16_rne(float f) {
  unsigned int u = __float_as_uint(f);
  u += 0x7FFFu + ((u >> 16) & 1u);
  return (unsigned short)(u >> 16);
}
static __device__ __forceinline__ float bf16f(unsigned short s) {
  return __uint_as_float(((unsigned int)s) << 16);
}

// ---------- fragment loaders ----------
// A (16x32 bf16, MxK): lane holds row M=lane&15; elements e0..15 -> K = kb + hi*16 + e.
// 16 contiguous bf16: two 16B chunks at p and p+8 (elements).
static __device__ __forceinline__ v16bf load_fragA(const unsigned short* p) {
  v16bf r;
  ((uint4*)&r)[0] = *(const uint4*)(p);
  ((uint4*)&r)[1] = *(const uint4*)(p + 8);
  return r;
}
// B (32x16 bf16, KxN): lane holds col N=lane&15; elements e0..15 -> K = kb + hi*16 + e.
// 16 contiguous bf16: two 16B chunks at p and p+8.
static __device__ __forceinline__ v16bf load_fragB(const unsigned short* p) {
  v16bf r;
  ((uint4*)&r)[0] = *(const uint4*)(p);
  ((uint4*)&r)[1] = *(const uint4*)(p + 8);
  return r;
}

static __device__ __forceinline__ v8f wmma_bf16(v16bf a, v16bf b, v8f c) {
  return __builtin_amdgcn_wmma_f32_16x16x32_bf16(false, a, false, b, (short)0, c,
                                                 false, false);
}

static __device__ __forceinline__ float sigf(float v) {
  return 1.0f / (1.0f + __expf(-v));
}
static __device__ __forceinline__ float tanh_fast(float v) {
  float e = __expf(2.0f * v);
  return 1.0f - 2.0f / (e + 1.0f);   // correct saturation at +-inf
}

// ---------- weight pre-split: fp32 -> bf16 hi/lo, combined [i;o;u;f] layout ----------
__global__ void td_split_weights(const float* __restrict__ W_iou,
                                 const float* __restrict__ W_f,
                                 const float* __restrict__ U_iou,
                                 const float* __restrict__ U_f,
                                 unsigned short* __restrict__ Wh,
                                 unsigned short* __restrict__ Wl,
                                 unsigned short* __restrict__ Uh,
                                 unsigned short* __restrict__ Ul) {
  int idx = blockIdx.x * blockDim.x + threadIdx.x;
  const int total = WROWS * TD_D;
  if (idx >= total) return;
  int r = idx / TD_D;
  int k = idx - r * TD_D;
  float w, u;
  if (r < 3 * TD_D) { w = W_iou[idx]; u = U_iou[idx]; }
  else {
    int rr = (r - 3 * TD_D) * TD_D + k;
    w = W_f[rr]; u = U_f[rr];
  }
  unsigned short wh = bf16_rne(w);
  Wh[idx] = wh;
  Wl[idx] = bf16_rne(w - bf16f(wh));
  unsigned short uh = bf16_rne(u);
  Uh[idx] = uh;
  Ul[idx] = bf16_rne(u - bf16f(uh));
}

// ---------- fused per-level kernel: GEMM (bf16x3 WMMA) + gates ----------
// Block: 32 nodes (2 m-tiles), 512 threads = 16 waves.
// Wave w computes i,o,u,f 16x16 tiles for 3 column tiles d0 = 3w..3w+2.
__global__ __launch_bounds__(512) void td_tree_level(
    const float* __restrict__ x,
    const float* __restrict__ b_iou,
    const float* __restrict__ b_f,
    const unsigned short* __restrict__ Wh,
    const unsigned short* __restrict__ Wl,
    const unsigned short* __restrict__ Uh,
    const unsigned short* __restrict__ Ul,
    float* __restrict__ outH,
    float* __restrict__ outC,
    int start, int width, int hasParent) {
  __shared__ unsigned short sAh[32 * LDS_STRIDE];
  __shared__ unsigned short sAl[32 * LDS_STRIDE];

  const int tid  = threadIdx.x;
  const int lane = tid & 31;
  const int wave = tid >> 5;       // 0..15
  const int n16  = lane & 15;
  const int khi  = lane >> 4;      // 0/1: K-half selector
  const int nodeBase = blockIdx.x * 32;

  for (int j = 0; j < 3; ++j) {
    const int d0 = wave * 3 + j;          // 0..47 column tile
    const int d  = d0 * 16 + n16;         // 0..767 feature index for this lane

    v8f acc[4][2];
    #pragma unroll
    for (int g = 0; g < 4; ++g)
      #pragma unroll
      for (int mt = 0; mt < 2; ++mt)
        acc[g][mt] = (v8f){0.f, 0.f, 0.f, 0.f, 0.f, 0.f, 0.f, 0.f};

    // ---- stage x rows -> LDS as bf16 hi/lo ----
    __syncthreads();
    for (int idx = tid; idx < 32 * TD_D; idx += 512) {
      int m = idx / TD_D;
      int k = idx - m * TD_D;
      int ml = nodeBase + m;
      float v = 0.f;
      if (ml < width) v = x[(size_t)(start + ml) * TD_D + k];
      unsigned short hb = bf16_rne(v);
      sAh[m * LDS_STRIDE + k] = hb;
      sAl[m * LDS_STRIDE + k] = bf16_rne(v - bf16f(hb));
    }
    __syncthreads();

    // ---- W pass: acc[g] += x_tile @ W_g^T  (bf16x3) ----
    for (int kc = 0; kc < TD_D / 32; ++kc) {
      const int kb = kc * 32;
      v16bf Bh[4], Bl[4];
      #pragma unroll
      for (int g = 0; g < 4; ++g) {
        size_t boff = (size_t)(g * TD_D + d0 * 16 + n16) * TD_D + kb + khi * 16;
        Bh[g] = load_fragB(Wh + boff);
        Bl[g] = load_fragB(Wl + boff);
      }
      #pragma unroll
      for (int mt = 0; mt < 2; ++mt) {
        int aoff = (mt * 16 + n16) * LDS_STRIDE + kb + khi * 16;
        v16bf Ah = load_fragA(sAh + aoff);
        v16bf Al = load_fragA(sAl + aoff);
        #pragma unroll
        for (int g = 0; g < 4; ++g) {
          acc[g][mt] = wmma_bf16(Ah, Bh[g], acc[g][mt]);
          acc[g][mt] = wmma_bf16(Ah, Bl[g], acc[g][mt]);
          acc[g][mt] = wmma_bf16(Al, Bh[g], acc[g][mt]);
        }
      }
    }

    if (hasParent) {
      // ---- stage parent-h rows -> LDS (gather via (node-1)>>1) ----
      __syncthreads();
      for (int idx = tid; idx < 32 * TD_D; idx += 512) {
        int m = idx / TD_D;
        int k = idx - m * TD_D;
        int ml = nodeBase + m;
        float v = 0.f;
        if (ml < width) {
          int parent = (start + ml - 1) >> 1;
          v = outH[(size_t)parent * TD_D + k];
        }
        unsigned short hb = bf16_rne(v);
        sAh[m * LDS_STRIDE + k] = hb;
        sAl[m * LDS_STRIDE + k] = bf16_rne(v - bf16f(hb));
      }
      __syncthreads();

      // ---- U pass: acc[g] += ph_tile @ U_g^T ----
      for (int kc = 0; kc < TD_D / 32; ++kc) {
        const int kb = kc * 32;
        v16bf Bh[4], Bl[4];
        #pragma unroll
        for (int g = 0; g < 4; ++g) {
          size_t boff = (size_t)(g * TD_D + d0 * 16 + n16) * TD_D + kb + khi * 16;
          Bh[g] = load_fragB(Uh + boff);
          Bl[g] = load_fragB(Ul + boff);
        }
        #pragma unroll
        for (int mt = 0; mt < 2; ++mt) {
          int aoff = (mt * 16 + n16) * LDS_STRIDE + kb + khi * 16;
          v16bf Ah = load_fragA(sAh + aoff);
          v16bf Al = load_fragA(sAl + aoff);
          #pragma unroll
          for (int g = 0; g < 4; ++g) {
            acc[g][mt] = wmma_bf16(Ah, Bh[g], acc[g][mt]);
            acc[g][mt] = wmma_bf16(Ah, Bl[g], acc[g][mt]);
            acc[g][mt] = wmma_bf16(Al, Bh[g], acc[g][mt]);
          }
        }
      }
    }

    // ---- fused gate math + store (C/D layout: element r -> M = khi*8 + r) ----
    float bI = b_iou[d];
    float bO = b_iou[TD_D + d];
    float bU = b_iou[2 * TD_D + d];
    float bF = b_f[d];
    #pragma unroll
    for (int mt = 0; mt < 2; ++mt) {
      #pragma unroll
      for (int r = 0; r < 8; ++r) {
        int ml = nodeBase + mt * 16 + khi * 8 + r;
        bool valid = (ml < width);
        int node = start + ml;
        float iv = acc[0][mt][r] + bI;
        float ov = acc[1][mt][r] + bO;
        float uv = acc[2][mt][r] + bU;
        float cc;
        if (hasParent) {
          float pc = 0.f;
          if (valid) pc = outC[(size_t)((node - 1) >> 1) * TD_D + d];
          float fv = acc[3][mt][r] + bF;
          cc = sigf(iv) * tanh_fast(uv) + sigf(fv) * pc;
        } else {
          cc = sigf(iv) * tanh_fast(uv);
        }
        float hh = sigf(ov) * tanh_fast(cc);
        if (valid) {
          outH[(size_t)node * TD_D + d] = hh;
          outC[(size_t)node * TD_D + d] = cc;
        }
      }
    }
  }
}

extern "C" void kernel_launch(void* const* d_in, const int* in_sizes, int n_in,
                              void* d_out, int out_size, void* d_ws, size_t ws_size,
                              hipStream_t stream) {
  const float* x     = (const float*)d_in[0];
  const float* W_iou = (const float*)d_in[1];
  const float* b_iou = (const float*)d_in[2];
  const float* U_iou = (const float*)d_in[3];
  const float* W_f   = (const float*)d_in[4];
  const float* b_f   = (const float*)d_in[5];
  const float* U_f   = (const float*)d_in[6];

  float* outH = (float*)d_out;                       // h: [N, D]
  float* outC = outH + (size_t)TD_N * TD_D;          // c: [N, D]

  // workspace: 4 bf16 arrays of [3072 x 768] = 18.9 MB total
  unsigned short* Wh = (unsigned short*)d_ws;
  const size_t mElems = (size_t)WROWS * TD_D;
  unsigned short* Wl = Wh + mElems;
  unsigned short* Uh = Wh + 2 * mElems;
  unsigned short* Ul = Wh + 3 * mElems;

  {
    int total = WROWS * TD_D;
    int threads = 256;
    int blocks = (total + threads - 1) / threads;
    td_split_weights<<<blocks, threads, 0, stream>>>(W_iou, W_f, U_iou, U_f,
                                                     Wh, Wl, Uh, Ul);
  }

  int start = 0;
  for (int lvl = 0; lvl < TD_LEVELS; ++lvl) {
    int width = 1 << lvl;
    int blocks = (width + 31) / 32;
    td_tree_level<<<blocks, 512, 0, stream>>>(x, b_iou, b_f, Wh, Wl, Uh, Ul,
                                              outH, outC, start, width,
                                              lvl > 0 ? 1 : 0);
    start += width;
  }
}
